// SparseMoEBlock_22574348108131
// MI455X (gfx1250) — matrix operand order repaired and checked by reading, not compiled, not run on hardware
//
#include <hip/hip_runtime.h>
#include <hip/hip_bf16.h>

#define T_TOK 4096
#define DIM   1024
#define HID   2048
#define NEXP  8

// Own trivially-copyable vector types (safe in unions / bit reinterpretation).
typedef unsigned int u32x4 __attribute__((ext_vector_type(4)));
typedef unsigned int u32x2 __attribute__((ext_vector_type(2)));
typedef float        f32x4 __attribute__((ext_vector_type(4)));
typedef float        f32x2 __attribute__((ext_vector_type(2)));
typedef __bf16       v16bf __attribute__((ext_vector_type(16)));
typedef __bf16       v2bf  __attribute__((ext_vector_type(2)));
typedef float        v8f   __attribute__((ext_vector_type(8)));

union Frag32B { u32x4 q[2]; v16bf v; };   // 32 bytes = 16 bf16 = one WMMA A/B operand

#define ASTR 40   // LDS row stride (bf16 elems) for 32-wide K tiles: 80B, 16B aligned

// LDS byte offset of a __shared__ element: local->flat cast keeps the LDS
// offset in the low 32 bits on AMDGPU.
#define LDS_OFF(p) ((unsigned)(unsigned long long)(const void*)(p))

// Async copy of 16B (and a second 16B at +16) global -> LDS, per lane.
// offset applies to BOTH the global and LDS address (ISA 10.x async ops).
__device__ __forceinline__ void async_copy_32B(unsigned lds_off, const void* gsrc) {
  asm volatile("global_load_async_to_lds_b128 %0, %1, off"
               :: "v"(lds_off), "v"(gsrc) : "memory");
  asm volatile("global_load_async_to_lds_b128 %0, %1, off offset:16"
               :: "v"(lds_off), "v"(gsrc) : "memory");
}
__device__ __forceinline__ void async_wait0() {
  asm volatile("s_wait_asynccnt 0x0" ::: "memory");
}

__device__ __forceinline__ float silu_fast(float g) {
  // g * rcp(1 + exp(-g)); rcp is fine at bf16 output precision.
  return g * __builtin_amdgcn_rcpf(1.f + __expf(-g));
}

// ---------------------------------------------------------------- router
// One wave (32 lanes) per token: logits = x_row @ Wr, top-2, softmax over top-2.
__global__ __launch_bounds__(128) void moe_router(const float* __restrict__ x,
                                                  const float* __restrict__ Wr,
                                                  float* __restrict__ comb) {
  int t = blockIdx.x * 4 + (threadIdx.x >> 5);
  int lane = threadIdx.x & 31;
  float p[NEXP];
#pragma unroll
  for (int e = 0; e < NEXP; ++e) p[e] = 0.f;
  const float* xr = x + (size_t)t * DIM;
  for (int d = lane; d < DIM; d += 32) {
    float xv = xr[d];
#pragma unroll
    for (int e = 0; e < NEXP; ++e) p[e] += xv * Wr[d * NEXP + e];
  }
#pragma unroll
  for (int e = 0; e < NEXP; ++e) {
#pragma unroll
    for (int off = 16; off > 0; off >>= 1) p[e] += __shfl_xor(p[e], off, 32);
  }
  if (lane == 0) {
    int e0 = 0; float v0 = p[0];
#pragma unroll
    for (int e = 1; e < NEXP; ++e) if (p[e] > v0) { v0 = p[e]; e0 = e; }
    int e1 = (e0 == 0) ? 1 : 0; float v1 = p[e1];
#pragma unroll
    for (int e = 0; e < NEXP; ++e)
      if (e != e0 && p[e] > v1) { v1 = p[e]; e1 = e; }
    float w0 = 1.f / (1.f + __expf(v1 - v0));   // softmax over {v0, v1}
    float w1 = 1.f - w0;
    float* c = comb + (size_t)t * NEXP;
#pragma unroll
    for (int e = 0; e < NEXP; ++e) c[e] = 0.f;
    c[e0] = w0; c[e1] = w1;
  }
}

// ---------------------------------------------------------------- x -> bf16
__global__ __launch_bounds__(256) void moe_cvt_x(const float* __restrict__ x,
                                                 void* __restrict__ xb_) {
  __bf16* xb = (__bf16*)xb_;
  size_t i = ((size_t)blockIdx.x * blockDim.x + threadIdx.x) * 4;
  f32x4 f = *(const f32x4*)(x + i);
  f32x2 lo, hi; lo.x = f.x; lo.y = f.y; hi.x = f.z; hi.y = f.w;
  union { u32x2 u; v2bf b[2]; } o;
  o.b[0] = __builtin_convertvector(lo, v2bf);
  o.b[1] = __builtin_convertvector(hi, v2bf);
  *(u32x2*)(xb + i) = o.u;
}

// ---------------------------------------------------------------- gate/up GEMM
// Block tile: 128 tokens x 64 hidden cols; K=DIM in 32-steps, double-buffered
// LDS (single barrier per step; async A copy + raw B loads overlap the WMMAs).
// 8 waves, each owns a 64x16 strip; computes BOTH gate and up (shared A frags),
// then silu(g)*u*comb -> Act (bf16).
__global__ __launch_bounds__(256)
void moe_gateup(const void* __restrict__ xb_, const float* __restrict__ Wg,
                const float* __restrict__ Wu, const float* __restrict__ comb,
                int e, void* __restrict__ act_) {
  const __bf16* xb = (const __bf16*)xb_;
  __bf16* act = (__bf16*)act_;

  __shared__ __bf16 As[2][128 * ASTR];   // A tiles, row-major [m][k]
  __shared__ __bf16 BgS[2][64 * ASTR];   // B tiles, col-major [n][k]
  __shared__ __bf16 BuS[2][64 * ASTR];
  __shared__ float  combS[128];

  const int tid = threadIdx.x;
  const int m0 = blockIdx.x * 128;
  const int n0 = blockIdx.y * 64;
  const int w = tid >> 5, lane = tid & 31;
  const int wm = w >> 2, wn = w & 3;          // 2 x 4 wave grid
  const int lhalf = lane >> 4, l16 = lane & 15;

  if (tid < 128) combS[tid] = comb[(size_t)(m0 + tid) * NEXP + e];

  v8f accG[4], accU[4];
#pragma unroll
  for (int i = 0; i < 4; ++i)
#pragma unroll
    for (int r = 0; r < 8; ++r) { accG[i][r] = 0.f; accU[i][r] = 0.f; }

  // A staging (async bf16 copy): thread -> (row, 32B half)
  const int arow = tid >> 1;            // 0..127
  const int acol = (tid & 1) * 16;      // 0 / 16
  const unsigned aLds[2] = { LDS_OFF(&As[0][arow * ASTR + acol]),
                             LDS_OFF(&As[1][arow * ASTR + acol]) };
  const __bf16* aSrc = xb + (size_t)(m0 + arow) * DIM + acol;
  // B staging (f32 -> packed bf16): thread -> (k-pair, 4 n cols)
  const int kp  = tid >> 4;             // 0..15 -> k = 2*kp, 2*kp+1
  const int nc  = (tid & 15) * 4;       // 0..60
  const float* gSrc = Wg + (size_t)(2 * kp) * HID + n0 + nc;
  const float* uSrc = Wu + (size_t)(2 * kp) * HID + n0 + nc;

  // ---- prologue: stage buffer 0 for kk = 0
  async_copy_32B(aLds[0], aSrc);
  {
    f32x4 g0 = *(const f32x4*)gSrc, g1 = *(const f32x4*)(gSrc + HID);
    f32x4 u0 = *(const f32x4*)uSrc, u1 = *(const f32x4*)(uSrc + HID);
#pragma unroll
    for (int j = 0; j < 4; ++j) {
      f32x2 fg; fg.x = g0[j]; fg.y = g1[j];
      f32x2 fu; fu.x = u0[j]; fu.y = u1[j];
      *(v2bf*)(&BgS[0][(nc + j) * ASTR + 2 * kp]) = __builtin_convertvector(fg, v2bf);
      *(v2bf*)(&BuS[0][(nc + j) * ASTR + 2 * kp]) = __builtin_convertvector(fu, v2bf);
    }
  }
  async_wait0();
  __syncthreads();

  const int NIT = DIM / 32;
  for (int it = 0; it < NIT; ++it) {
    const int cur = it & 1, nxt = cur ^ 1;
    const int kk = it * 32;
    const bool hasNext = (it + 1) < NIT;

    // ---- issue next tile's loads up front (overlap with WMMAs below)
    f32x4 g0, g1, u0, u1;
    if (hasNext) {
      const float* gp = gSrc + (size_t)(kk + 32) * HID;
      const float* up = uSrc + (size_t)(kk + 32) * HID;
      g0 = *(const f32x4*)gp; g1 = *(const f32x4*)(gp + HID);
      u0 = *(const f32x4*)up; u1 = *(const f32x4*)(up + HID);
      async_copy_32B(aLds[nxt], aSrc + kk + 32);
    }

    // ---- fragments from current buffer (CDNA5 16-bit layouts, §7.12.2)
    const __bf16* Ab = As[cur];
    const __bf16* Gb = BgS[cur];
    const __bf16* Ub = BuS[cur];
    Frag32B a[4], bg, bu;
#pragma unroll
    for (int i = 0; i < 4; ++i) {
      int row = wm * 64 + i * 16 + l16;
      // lanes 0-15: K 0-7 / 16-23 ; lanes 16-31: K 8-15 / 24-31
      a[i].q[0] = *(const u32x4*)(&Ab[row * ASTR + lhalf * 8]);
      a[i].q[1] = *(const u32x4*)(&Ab[row * ASTR + 16 + lhalf * 8]);
    }
    {
      int col = wn * 16 + l16;
      // lane holds column `col`; lanes 0-15: K 0-7 / 16-23 ; lanes 16-31: K 8-15 / 24-31
      bg.q[0] = *(const u32x4*)(&Gb[col * ASTR + lhalf * 8]);
      bg.q[1] = *(const u32x4*)(&Gb[col * ASTR + 16 + lhalf * 8]);
      bu.q[0] = *(const u32x4*)(&Ub[col * ASTR + lhalf * 8]);
      bu.q[1] = *(const u32x4*)(&Ub[col * ASTR + 16 + lhalf * 8]);
    }
#pragma unroll
    for (int i = 0; i < 4; ++i) {
      accG[i] = __builtin_amdgcn_wmma_f32_16x16x32_bf16(
          false, a[i].v, false, bg.v, (short)0, accG[i], false, false);
      accU[i] = __builtin_amdgcn_wmma_f32_16x16x32_bf16(
          false, a[i].v, false, bu.v, (short)0, accU[i], false, false);
    }

    // ---- finish staging next buffer (pack + LDS stores), one barrier/step
    if (hasNext) {
#pragma unroll
      for (int j = 0; j < 4; ++j) {
        f32x2 fg; fg.x = g0[j]; fg.y = g1[j];
        f32x2 fu; fu.x = u0[j]; fu.y = u1[j];
        *(v2bf*)(&BgS[nxt][(nc + j) * ASTR + 2 * kp]) = __builtin_convertvector(fg, v2bf);
        *(v2bf*)(&BuS[nxt][(nc + j) * ASTR + 2 * kp]) = __builtin_convertvector(fu, v2bf);
      }
    }
    async_wait0();
    __syncthreads();
  }

  // ---- epilogue: silu(g) * u * comb[token,e] -> Act (bf16)
  // C/D layout: VGPR r holds M = r + 8*(lane>=16), N = lane&15.
#pragma unroll
  for (int i = 0; i < 4; ++i) {
    int mrel = wm * 64 + i * 16 + lhalf * 8;
    int ng = n0 + wn * 16 + l16;
#pragma unroll
    for (int r = 0; r < 8; ++r) {
      float av = silu_fast(accG[i][r]) * accU[i][r] * combS[mrel + r];
      act[(size_t)(m0 + mrel + r) * HID + ng] = (__bf16)av;
    }
  }
}

// ---------------------------------------------------------------- down GEMM
// out[128 x 64 tile] += Act[128, HID] @ Wd[HID, 64 cols]. Sequential expert
// kernels on the stream make the global read-modify-write race-free.
__global__ __launch_bounds__(256)
void moe_down(const void* __restrict__ act_, const float* __restrict__ Wd,
              float* __restrict__ out) {
  const __bf16* act = (const __bf16*)act_;
  __shared__ __bf16 As[2][128 * ASTR];
  __shared__ __bf16 BS[2][64 * ASTR];

  const int tid = threadIdx.x;
  const int m0 = blockIdx.x * 128;
  const int n0 = blockIdx.y * 64;
  const int w = tid >> 5, lane = tid & 31;
  const int wm = w >> 2, wn = w & 3;
  const int lhalf = lane >> 4, l16 = lane & 15;

  v8f acc[4];
#pragma unroll
  for (int i = 0; i < 4; ++i)
#pragma unroll
    for (int r = 0; r < 8; ++r) acc[i][r] = 0.f;

  const int arow = tid >> 1;
  const int acol = (tid & 1) * 16;
  const unsigned aLds[2] = { LDS_OFF(&As[0][arow * ASTR + acol]),
                             LDS_OFF(&As[1][arow * ASTR + acol]) };
  const __bf16* aSrc = act + (size_t)(m0 + arow) * HID + acol;
  const int kp  = tid >> 4;
  const int nc  = (tid & 15) * 4;
  const float* dSrc = Wd + (size_t)(2 * kp) * DIM + n0 + nc;

  // ---- prologue: stage buffer 0
  async_copy_32B(aLds[0], aSrc);
  {
    f32x4 d0 = *(const f32x4*)dSrc, d1 = *(const f32x4*)(dSrc + DIM);
#pragma unroll
    for (int j = 0; j < 4; ++j) {
      f32x2 fd; fd.x = d0[j]; fd.y = d1[j];
      *(v2bf*)(&BS[0][(nc + j) * ASTR + 2 * kp]) = __builtin_convertvector(fd, v2bf);
    }
  }
  async_wait0();
  __syncthreads();

  const int NIT = HID / 32;
  for (int it = 0; it < NIT; ++it) {
    const int cur = it & 1, nxt = cur ^ 1;
    const int kk = it * 32;
    const bool hasNext = (it + 1) < NIT;

    f32x4 d0, d1;
    if (hasNext) {
      const float* dp = dSrc + (size_t)(kk + 32) * DIM;
      d0 = *(const f32x4*)dp; d1 = *(const f32x4*)(dp + DIM);
      async_copy_32B(aLds[nxt], aSrc + kk + 32);
    }

    const __bf16* Ab = As[cur];
    const __bf16* Bb = BS[cur];
    Frag32B a[4], b;
#pragma unroll
    for (int i = 0; i < 4; ++i) {
      int row = wm * 64 + i * 16 + l16;
      a[i].q[0] = *(const u32x4*)(&Ab[row * ASTR + lhalf * 8]);
      a[i].q[1] = *(const u32x4*)(&Ab[row * ASTR + 16 + lhalf * 8]);
    }
    {
      int col = wn * 16 + l16;
      b.q[0] = *(const u32x4*)(&Bb[col * ASTR + lhalf * 8]);
      b.q[1] = *(const u32x4*)(&Bb[col * ASTR + 16 + lhalf * 8]);
    }
#pragma unroll
    for (int i = 0; i < 4; ++i) {
      acc[i] = __builtin_amdgcn_wmma_f32_16x16x32_bf16(
          false, a[i].v, false, b.v, (short)0, acc[i], false, false);
    }

    if (hasNext) {
#pragma unroll
      for (int j = 0; j < 4; ++j) {
        f32x2 fd; fd.x = d0[j]; fd.y = d1[j];
        *(v2bf*)(&BS[nxt][(nc + j) * ASTR + 2 * kp]) = __builtin_convertvector(fd, v2bf);
      }
    }
    async_wait0();
    __syncthreads();
  }

#pragma unroll
  for (int i = 0; i < 4; ++i) {
    int mrel = wm * 64 + i * 16 + lhalf * 8;
    int ng = n0 + wn * 16 + l16;
#pragma unroll
    for (int r = 0; r < 8; ++r) {
      float* o = out + (size_t)(m0 + mrel + r) * DIM + ng;
      *o += acc[i][r];
    }
  }
}

// ---------------------------------------------------------------- launch
extern "C" void kernel_launch(void* const* d_in, const int* in_sizes, int n_in,
                              void* d_out, int out_size, void* d_ws, size_t ws_size,
                              hipStream_t stream) {
  (void)in_sizes; (void)n_in; (void)ws_size;
  const float* x  = (const float*)d_in[0];
  const float* Wr = (const float*)d_in[1];
  const float* Wg = (const float*)d_in[2];
  const float* Wu = (const float*)d_in[3];
  const float* Wd = (const float*)d_in[4];
  float* out = (float*)d_out;

  char* ws = (char*)d_ws;
  float* comb = (float*)ws;                                        // 128 KB
  void*  xb   = (void*)(ws + (256 << 10));                         // 8 MB bf16 x
  void*  act  = (void*)(ws + (256 << 10) + ((size_t)8 << 20));     // 16 MB bf16 Act

  hipMemsetAsync(d_out, 0, (size_t)out_size * sizeof(float), stream);

  moe_router<<<T_TOK / 4, 128, 0, stream>>>(x, Wr, comb);
  moe_cvt_x<<<(T_TOK * DIM) / (256 * 4), 256, 0, stream>>>(x, xb);

  for (int e = 0; e < NEXP; ++e) {
    moe_gateup<<<dim3(T_TOK / 128, HID / 64), 256, 0, stream>>>(
        xb, Wg + (size_t)e * DIM * HID, Wu + (size_t)e * DIM * HID, comb, e, act);
    moe_down<<<dim3(T_TOK / 128, DIM / 64), 256, 0, stream>>>(
        act, Wd + (size_t)e * HID * DIM, out);
  }
}
